// LSTM_22308060135990
// MI455X (gfx1250) — matrix operand order repaired and checked by reading, not compiled, not run on hardware
//
#include <hip/hip_runtime.h>
#include <hip/hip_bf16.h>

// LSTM: T=512, B=256, D=H=1024, L=2, gates = 4H = 4096
#define LSTM_T  512
#define LSTM_B  256
#define LSTM_D  1024
#define LSTM_H  1024
#define LSTM_NG 4096

// GEMM tiling: per-launch M=256, N=4096, K=2048 (x-path + h-path)
#define BM     64
#define BN     128
#define KB     32    // one v_wmma_f32_16x16x32_bf16 k-depth
#define LDSB   40    // halves per LDS row (32 + 8 pad), 80B keeps 16B alignment
#define KSTEPS 64    // 2 sources * (1024/32)

typedef __attribute__((ext_vector_type(16))) __bf16 v16bf;
typedef __attribute__((ext_vector_type(8)))  __bf16 v8bf;
typedef __attribute__((ext_vector_type(8)))  float  v8f;

__device__ __forceinline__ unsigned short f32_to_bf16(float f) {
  unsigned u = __float_as_uint(f);
  return (unsigned short)((u + 0x7FFFu + ((u >> 16) & 1u)) >> 16);  // RNE
}

// ---------------- fp32 -> bf16 bulk conversion (weights + x_0) ---------------
__global__ __launch_bounds__(256) void cvt_bf16_kernel(const float* __restrict__ src,
                                                       unsigned short* __restrict__ dst,
                                                       int n) {
  int i = blockIdx.x * blockDim.x + threadIdx.x;
  if (i < n) dst[i] = f32_to_bf16(src[i]);
}

__global__ __launch_bounds__(256) void zero_f32_kernel(float* __restrict__ p, int n) {
  int i = blockIdx.x * blockDim.x + threadIdx.x;
  if (i < n) p[i] = 0.0f;
}

__global__ __launch_bounds__(256) void zero_u16_kernel(unsigned short* __restrict__ p,
                                                       int n) {
  int i = blockIdx.x * blockDim.x + threadIdx.x;
  if (i < n) p[i] = 0;
}

// ---------------- recurrent GEMM: comb = A0@W0^T + A1@W1^T -------------------
// A0,A1: [256,1024] bf16 activations (x_t and h_prev, pre-converted)
// W0,W1: [4096,1024] bf16 weights (row-major, K contiguous)
// comb : [256,4096] fp32
// Block = 64x128 tile, 8 waves, each wave owns a 32x32 patch (2x2 WMMA tiles).
// LDS double-buffered: one barrier per k-step; staging is pure b128 copies.
__global__ __launch_bounds__(256) void lstm_gemm_kernel(
    const unsigned short* __restrict__ A0, const unsigned short* __restrict__ A1,
    const unsigned short* __restrict__ W0, const unsigned short* __restrict__ W1,
    float* __restrict__ comb)
{
  __shared__ unsigned short As[2][BM * LDSB];   // activation tiles (bf16)
  __shared__ unsigned short Bs[2][BN * LDSB];   // weight tiles (bf16)

  const int tid  = threadIdx.x;
  const int lane = tid & 31;
  const int wave = tid >> 5;
  const int wm   = (wave & 1) * 32;    // 2 wave-rows  x 32 = 64  (BM)
  const int wn   = (wave >> 1) * 32;   // 4 wave-cols  x 32 = 128 (BN)
  const int bm   = blockIdx.x * BM;
  const int bn   = blockIdx.y * BN;

  v8f acc[2][2] = {};                  // 2x2 16x16 f32 accumulators per wave

  // cooperative staging mapping (pure copies, 8/16 halves per thread)
  const int alr = tid >> 2;            // A: 4 threads/row, 8 bf16 each
  const int alc = (tid & 3) * 8;
  const int blr = tid >> 1;            // B: 2 threads/row, 16 bf16 each
  const int blc = (tid & 1) * 16;

  const int l16 = lane & 15;
  const int ka  = (lane < 16) ? 0 : 8;    // A frag: K {0..7,16..23} vs {8..15,24..31}
  const int kb  = (lane < 16) ? 0 : 8;    // B frag: K {0..7,16..23} vs {8..15,24..31}

  // stage k-step kk (0..63; kk>>5 selects source, kk&31 selects k-chunk)
  auto stage = [&](int buf, int kk) {
    const int src = kk >> 5;
    const int k0  = (kk & 31) * KB;
    const unsigned short* ap =
        (src ? A1 : A0) + (size_t)(bm + alr) * LSTM_D + (k0 + alc);
    *(uint4*)(&As[buf][alr * LDSB + alc]) = *(const uint4*)(ap);
    const unsigned short* wp =
        (src ? W1 : W0) + (size_t)(bn + blr) * LSTM_D + (k0 + blc);
    *(uint4*)(&Bs[buf][blr * LDSB + blc])     = *(const uint4*)(wp);
    *(uint4*)(&Bs[buf][blr * LDSB + blc + 8]) = *(const uint4*)(wp + 8);
    __builtin_prefetch(wp + KB, 0, 1);     // next k-line of this weight row
  };

  stage(0, 0);
  int cur = 0;

  #pragma unroll 1
  for (int kk = 0; kk < KSTEPS; ++kk) {
    __syncthreads();                       // buf 'cur' staged & prior reads done
    if (kk + 1 < KSTEPS) stage(cur ^ 1, kk + 1);

    // A fragments: 16x32 bf16, lane = M row
    v16bf af[2];
    #pragma unroll
    for (int mi = 0; mi < 2; ++mi) {
      const int arow = wm + mi * 16 + l16;
      v8bf lo = *(const v8bf*)(&As[cur][arow * LDSB + ka]);
      v8bf hi = *(const v8bf*)(&As[cur][arow * LDSB + ka + 16]);
      af[mi] = __builtin_shufflevector(lo, hi,
          0,1,2,3,4,5,6,7,8,9,10,11,12,13,14,15);
    }
    // B fragments: 32x16 bf16, lane = N col (weight row)
    v16bf bfg[2];
    #pragma unroll
    for (int ni = 0; ni < 2; ++ni) {
      const int brow = wn + ni * 16 + l16;
      v8bf lo = *(const v8bf*)(&Bs[cur][brow * LDSB + kb]);
      v8bf hi = *(const v8bf*)(&Bs[cur][brow * LDSB + kb + 16]);
      bfg[ni] = __builtin_shufflevector(lo, hi,
          0,1,2,3,4,5,6,7,8,9,10,11,12,13,14,15);
    }
    #pragma unroll
    for (int mi = 0; mi < 2; ++mi)
      #pragma unroll
      for (int ni = 0; ni < 2; ++ni)
        acc[mi][ni] = __builtin_amdgcn_wmma_f32_16x16x32_bf16(
            false, af[mi], false, bfg[ni], (short)0, acc[mi][ni], false, false);

    cur ^= 1;
  }

  // writeback: C/D layout lanes 0-15 -> M=r, lanes 16-31 -> M=r+8
  // (lanes cover 16 consecutive columns -> stores coalesce to 64B lines)
  const int rhi = (lane < 16) ? 0 : 8;
  #pragma unroll
  for (int mi = 0; mi < 2; ++mi) {
    const int rowb = bm + wm + mi * 16 + rhi;
    #pragma unroll
    for (int ni = 0; ni < 2; ++ni) {
      const int col = bn + wn + ni * 16 + l16;
      #pragma unroll
      for (int r = 0; r < 8; ++r) {
        comb[(size_t)(rowb + r) * LSTM_NG + col] = acc[mi][ni][r];
      }
    }
  }
}

// ---------------- fused LSTM cell: bias + gates + state update ---------------
// Also emits bf16 h for the next GEMM, and (optionally) converts the next
// timestep's x slice to bf16 so the GEMM hot loop never touches fp32 inputs.
__global__ __launch_bounds__(256) void lstm_cell_kernel(
    const float* __restrict__ comb, const float* __restrict__ bias,
    float* __restrict__ h, float* __restrict__ c,
    unsigned short* __restrict__ hb,
    float* __restrict__ out,
    const float* __restrict__ x_next, unsigned short* __restrict__ xb_next)
{
  int i = blockIdx.x * blockDim.x + threadIdx.x;   // 0 .. B*H-1
  int b = i >> 10;
  int j = i & (LSTM_H - 1);
  const float* cb = comb + (size_t)b * LSTM_NG;
  float ig = cb[j]              + bias[j];
  float fg = cb[LSTM_H + j]     + bias[LSTM_H + j];
  float og = cb[2 * LSTM_H + j] + bias[2 * LSTM_H + j];
  float gg = cb[3 * LSTM_H + j] + bias[3 * LSTM_H + j];
  ig = 1.0f / (1.0f + __expf(-ig));
  fg = 1.0f / (1.0f + __expf(-fg));
  og = 1.0f / (1.0f + __expf(-og));
  gg = tanhf(gg);
  float cn = fg * c[i] + ig * gg;
  float hn = og * tanhf(cn);
  c[i]  = cn;
  h[i]  = hn;
  hb[i] = f32_to_bf16(hn);
  if (out) out[i] = hn;
  if (x_next) xb_next[i] = f32_to_bf16(x_next[i]);   // B*H == B*D elements
}

// -----------------------------------------------------------------------------
extern "C" void kernel_launch(void* const* d_in, const int* in_sizes, int n_in,
                              void* d_out, int out_size, void* d_ws, size_t ws_size,
                              hipStream_t stream) {
  const float* X      = (const float*)d_in[0];   // [T,B,D]
  const float* w_in0  = (const float*)d_in[1];   // [4H,D]
  const float* w_hid0 = (const float*)d_in[2];   // [4H,H]
  const float* b0     = (const float*)d_in[3];   // [4H]
  const float* w_in1  = (const float*)d_in[4];   // [4H,H]
  const float* w_hid1 = (const float*)d_in[5];   // [4H,H]
  const float* b1     = (const float*)d_in[6];   // [4H]
  float* out = (float*)d_out;

  // workspace: 4x bf16 weights (32MB) + comb (4MB) + fp32 h/c (4MB)
  //          + bf16 h (1MB) + bf16 x ping-pong (1MB)
  const size_t WE = (size_t)LSTM_NG * LSTM_D;        // weight elements
  const size_t SE = (size_t)LSTM_B * LSTM_H;         // state elements
  unsigned short* wb_in0  = (unsigned short*)d_ws;
  unsigned short* wb_hid0 = wb_in0  + WE;
  unsigned short* wb_in1  = wb_hid0 + WE;
  unsigned short* wb_hid1 = wb_in1  + WE;
  float* comb = (float*)(wb_hid1 + WE);
  float* h0   = comb + (size_t)LSTM_B * LSTM_NG;
  float* c0   = h0 + SE;
  float* h1   = c0 + SE;
  float* c1   = h1 + SE;
  unsigned short* h0b  = (unsigned short*)(c1 + SE);
  unsigned short* h1b  = h0b + SE;
  unsigned short* xb0  = h1b + SE;       // x_t bf16 ping-pong
  unsigned short* xb1  = xb0 + SE;
  unsigned short* xb[2] = { xb0, xb1 };

  // one-time (per call) weight conversion fp32 -> bf16
  {
    dim3 g((unsigned)((WE + 255) / 256));
    cvt_bf16_kernel<<<g, 256, 0, stream>>>(w_in0,  wb_in0,  (int)WE);
    cvt_bf16_kernel<<<g, 256, 0, stream>>>(w_hid0, wb_hid0, (int)WE);
    cvt_bf16_kernel<<<g, 256, 0, stream>>>(w_in1,  wb_in1,  (int)WE);
    cvt_bf16_kernel<<<g, 256, 0, stream>>>(w_hid1, wb_hid1, (int)WE);
  }
  // zero fp32 h/c (contiguous) and bf16 h
  {
    int nf = (int)(4 * SE);
    zero_f32_kernel<<<(nf + 255) / 256, 256, 0, stream>>>(h0, nf);
    int nu = (int)(2 * SE);
    zero_u16_kernel<<<(nu + 255) / 256, 256, 0, stream>>>(h0b, nu);
  }
  // convert x_0
  {
    int n = (int)SE;
    cvt_bf16_kernel<<<(n + 255) / 256, 256, 0, stream>>>(X, xb[0], n);
  }

  dim3 ggrid(LSTM_B / BM, LSTM_NG / BN);   // (4, 32) = 128 workgroups
  const int cell_blocks = (int)(SE / 256);

  for (int t = 0; t < LSTM_T; ++t) {
    const float* x_next = (t + 1 < LSTM_T) ? X + (size_t)(t + 1) * SE : nullptr;
    unsigned short* xb_next = (t + 1 < LSTM_T) ? xb[(t + 1) & 1] : nullptr;
    // layer 0
    lstm_gemm_kernel<<<ggrid, 256, 0, stream>>>(xb[t & 1], h0b, wb_in0, wb_hid0, comb);
    lstm_cell_kernel<<<cell_blocks, 256, 0, stream>>>(
        comb, b0, h0, c0, h0b, nullptr, x_next, xb_next);
    // layer 1 (input = h0 just produced; its output is the model output)
    lstm_gemm_kernel<<<ggrid, 256, 0, stream>>>(h0b, h1b, wb_in1, wb_hid1, comb);
    lstm_cell_kernel<<<cell_blocks, 256, 0, stream>>>(
        comb, b1, h1, c1, h1b, out + (size_t)t * SE, nullptr, nullptr);
  }

  // final states: hT = [h0, h1], cT = [c0, c1] appended after output
  const size_t SB = SE * sizeof(float);
  float* tail = out + (size_t)LSTM_T * SE;
  hipMemcpyAsync(tail,          h0, SB, hipMemcpyDeviceToDevice, stream);
  hipMemcpyAsync(tail + 1 * SE, h1, SB, hipMemcpyDeviceToDevice, stream);
  hipMemcpyAsync(tail + 2 * SE, c0, SB, hipMemcpyDeviceToDevice, stream);
  hipMemcpyAsync(tail + 3 * SE, c1, SB, hipMemcpyDeviceToDevice, stream);
}
